// AgentSelfAttention_73134703117016
// MI455X (gfx1250) — matrix operand order repaired and checked by reading, not compiled, not run on hardware
//
#include <hip/hip_runtime.h>

// ---------------------------------------------------------------------------
// Agent attention, CDNA5 (gfx1250, wave32) bf16-WMMA pipeline.
// B=4 N=4096 D=1024 H=16 M=128 DH=64.  QKV GEMM (103 GFLOP) dominates ->
// v_wmma_f32_16x16x32_bf16 with f32 accumulation.  Fragments use contiguous
// K order k = 16h + s in A and B (wave32).  L2 (192MB) carries cross-wave operand reuse;
// LDS used where a transpose or block-wide operand sharing pays:
//   * k_agent_gather: v-tile staged via GLOBAL_LOAD_ASYNC_TO_LDS_B128
//     (ASYNCcnt path, inline asm) then transposed in LDS.
//   * k_out_gemm:     agbT operand block DMA'd by the Tensor Data Mover
//     (TENSOR_LOAD_TO_LDS + s_wait_tensorcnt).
// ---------------------------------------------------------------------------

typedef __attribute__((ext_vector_type(16))) __bf16 v16bf;
typedef __attribute__((ext_vector_type(8)))  __bf16 v8bf;
typedef __attribute__((ext_vector_type(8)))  float  v8f;
typedef __attribute__((ext_vector_type(4)))  unsigned v4u;
typedef __attribute__((ext_vector_type(4)))  int      v4i32;
typedef __attribute__((ext_vector_type(8)))  int      v8i32;

constexpr int B  = 4;
constexpr int N  = 4096;
constexpr int D  = 1024;
constexpr int H  = 16;
constexpr int M  = 128;
constexpr int DH = 64;
constexpr int D3 = 3 * D;
constexpr int BN = B * N;
constexpr float SCALE = 0.125f;   // DH^-0.5

// A-matrix fragment (16x32 bf16), row-major source, row stride ld (elems).
// Layout used here (same K order as loadB): lane L holds row M=L&15; lanes<16 -> K 0..15,
// lanes>=16 -> K 16..31.  src points at (row0, k0).
__device__ __forceinline__ v16bf loadA(const __bf16* src, int ld, int lane) {
  const __bf16* base = src + (size_t)(lane & 15) * ld;
  int half = lane >> 4;
  v8bf lo = *(const v8bf*)(base + half * 16);
  v8bf hi = *(const v8bf*)(base + half * 16 + 8);
  return __builtin_shufflevector(lo, hi, 0,1,2,3,4,5,6,7,8,9,10,11,12,13,14,15);
}

// B-matrix fragment (32x16 bf16).  Memory holds column n contiguous along K
// (element (k, col) at src[col*ld + k]).  Lane = column (lane&15),
// lanes<16 -> K 0..15, lanes>=16 -> K 16..31 (contiguous 16 halves = 32B).
__device__ __forceinline__ v16bf loadB(const __bf16* src, int ld, int lane) {
  return *(const v16bf*)(src + (size_t)(lane & 15) * ld + (lane >> 4) * 16);
}

__device__ __forceinline__ v8f wmma_bf16(v16bf a, v16bf b, v8f c) {
  return __builtin_amdgcn_wmma_f32_16x16x32_bf16(false, a, false, b, (short)0, c,
                                                 false, false);
}

// Tensor Data Mover: load one tile described by D# groups 0/1 (2D case,
// groups 2/3 zero).  Builtin arity differs across toolchains.
__device__ __forceinline__ void tdm_load_to_lds(v4u g0, v8i32 g1) {
  v4i32 z4 = {0, 0, 0, 0};
#if defined(__clang_major__) && (__clang_major__ >= 23)
  v8i32 z8 = {0, 0, 0, 0, 0, 0, 0, 0};
  __builtin_amdgcn_tensor_load_to_lds(g0, g1, z4, z4, z8, 0);
#else
  __builtin_amdgcn_tensor_load_to_lds(g0, g1, z4, z4, 0);
#endif
}

// ---------------------------------------------------------------------------
// K1: fp32 -> bf16 conversion of x, w_fused, agents (pre-scaled); zero agf.
// ---------------------------------------------------------------------------
__global__ void __launch_bounds__(256)
k_convert(const float* __restrict__ x, const float* __restrict__ w,
          const float* __restrict__ ag, __bf16* __restrict__ xb,
          __bf16* __restrict__ wb, __bf16* __restrict__ ab,
          float* __restrict__ agf) {
  int stride = gridDim.x * blockDim.x;
  int t0 = blockIdx.x * blockDim.x + threadIdx.x;
  for (int i = t0; i < BN * D;     i += stride) xb[i] = (__bf16)x[i];
  for (int i = t0; i < D3 * D;     i += stride) wb[i] = (__bf16)w[i];
  for (int i = t0; i < H * M * DH; i += stride) ab[i] = (__bf16)(ag[i] * SCALE);
  for (int i = t0; i < B * H * DH * M; i += stride) agf[i] = 0.0f;
}

// ---------------------------------------------------------------------------
// K2: qkv[BN x 3072] = xb[BN x 1024] @ wb^T.  Both operands K-contiguous.
// Block = 8 waves; wave computes 16 rows x 64 cols.
// ---------------------------------------------------------------------------
__global__ void __launch_bounds__(256)
k_qkv(const __bf16* __restrict__ xb, const __bf16* __restrict__ wb,
      __bf16* __restrict__ qkvb) {
  int lane = threadIdx.x & 31, wave = threadIdx.x >> 5;
  int r0 = blockIdx.y * 128 + wave * 16;
  int c0 = blockIdx.x * 64;
  const __bf16* Arow = xb + (size_t)r0 * D;
  v8f acc[4] = {};
  for (int k = 0; k < D; k += 32) {
    __builtin_prefetch(Arow + k + 128, 0, 0);           // global_prefetch_b8
    v16bf a = loadA(Arow + k, D, lane);
#pragma unroll
    for (int j = 0; j < 4; ++j) {
      v16bf b = loadB(wb + (size_t)(c0 + j * 16) * D + k, D, lane);
      acc[j] = wmma_bf16(a, b, acc[j]);
    }
  }
#pragma unroll
  for (int j = 0; j < 4; ++j)
#pragma unroll
    for (int e = 0; e < 8; ++e) {
      int m = e + (lane >> 4) * 8;                      // C layout: VGPR e
      int c = c0 + j * 16 + (lane & 15);
      qkvb[(size_t)(r0 + m) * D3 + c] = (__bf16)acc[j][e];
    }
}

// ---------------------------------------------------------------------------
// K3: ak scores  S[b,h,m,n] = a[h,m,:] . k[b,n,:]   (K=DH=64)
// ---------------------------------------------------------------------------
__global__ void __launch_bounds__(256)
k_scores_ak(const __bf16* __restrict__ ab, const __bf16* __restrict__ qkvb,
            __bf16* __restrict__ Sk) {
  int lane = threadIdx.x & 31, wave = threadIdx.x >> 5;
  int wid = blockIdx.x * 8 + wave;          // B*H*(M/16)*(N/64) = 32768
  int n64 = wid & 63;
  int m16 = (wid >> 6) & 7;
  int h   = (wid >> 9) & 15;
  int b   = wid >> 13;
  int n0 = n64 * 64, m0 = m16 * 16;
  const __bf16* Arow  = ab + (size_t)(h * M + m0) * DH;
  const __bf16* Bbase = qkvb + (size_t)b * N * D3 + D + h * DH;  // k block
  v8f acc[4] = {};
  for (int k = 0; k < DH; k += 32) {
    v16bf a = loadA(Arow + k, DH, lane);
#pragma unroll
    for (int j = 0; j < 4; ++j) {
      v16bf bf = loadB(Bbase + (size_t)(n0 + j * 16) * D3 + k, D3, lane);
      acc[j] = wmma_bf16(a, bf, acc[j]);
    }
  }
  __bf16* dst = Sk + (size_t)(b * H + h) * M * N;
#pragma unroll
  for (int j = 0; j < 4; ++j)
#pragma unroll
    for (int e = 0; e < 8; ++e) {
      int m = m0 + e + (lane >> 4) * 8;
      int n = n0 + j * 16 + (lane & 15);
      dst[(size_t)m * N + n] = (__bf16)acc[j][e];
    }
}

// ---------------------------------------------------------------------------
// K4: row softmax stats (max, 1/sum) for rows of length rowLen.
// ---------------------------------------------------------------------------
__global__ void __launch_bounds__(256)
k_rowstats(const __bf16* __restrict__ S, float* __restrict__ stats, int rowLen) {
  __shared__ float red[256];
  const __bf16* p = S + (size_t)blockIdx.x * rowLen;
  int t = threadIdx.x;
  float mx = -1e30f;
  for (int i = t; i < rowLen; i += 256) mx = fmaxf(mx, (float)p[i]);
  red[t] = mx; __syncthreads();
  for (int s = 128; s > 0; s >>= 1) {
    if (t < s) red[t] = fmaxf(red[t], red[t + s]);
    __syncthreads();
  }
  mx = red[0]; __syncthreads();
  float sum = 0.0f;
  for (int i = t; i < rowLen; i += 256) sum += __expf((float)p[i] - mx);
  red[t] = sum; __syncthreads();
  for (int s = 128; s > 0; s >>= 1) {
    if (t < s) red[t] += red[t + s];
    __syncthreads();
  }
  if (t == 0) {
    stats[(size_t)blockIdx.x * 2]     = mx;
    stats[(size_t)blockIdx.x * 2 + 1] = 1.0f / red[0];
  }
}

// ---------------------------------------------------------------------------
// K5: softmax-normalize + talking-heads mix, IN PLACE.
// S viewed as [B][H][rowsPer][rowLen]; block owns one (b, r) across all 16
// heads, so read-all-h / write-all-g at identical addresses is race-free.
// ---------------------------------------------------------------------------
__global__ void __launch_bounds__(256)
k_mix(__bf16* __restrict__ S, const float* __restrict__ stats,
      const float* __restrict__ w, int rowsPer, int rowLen) {
  __shared__ float wsm[H * H];
  __shared__ float st[H][2];
  int t = threadIdx.x;
  int b = blockIdx.x / rowsPer;
  int r = blockIdx.x - b * rowsPer;
  if (t < H * H) wsm[t] = w[t];
  if (t < H) {
    size_t srow = ((size_t)(b * H + t) * rowsPer + r) * 2;
    st[t][0] = stats[srow];
    st[t][1] = stats[srow + 1];
  }
  __syncthreads();
  size_t planeStride = (size_t)rowsPer * rowLen;
  size_t base = (size_t)b * H * planeStride + (size_t)r * rowLen;
  for (int n = t; n < rowLen; n += 256) {
    float pv[H];
#pragma unroll
    for (int h = 0; h < H; ++h) {
      float s = (float)S[base + (size_t)h * planeStride + n];
      pv[h] = __expf(s - st[h][0]) * st[h][1];
    }
#pragma unroll
    for (int g = 0; g < H; ++g) {
      float acc = 0.0f;
#pragma unroll
      for (int h = 0; h < H; ++h) acc += wsm[g * H + h] * pv[h];
      S[base + (size_t)g * planeStride + n] = (__bf16)acc;
    }
  }
}

// ---------------------------------------------------------------------------
// K6: agent_gathered^T[b,g,d,m] += mixed_ak[b,g,m,:] @ v[b,:,g,:]
// K (=N) split over gridDim.y, partial sums via global f32 atomics.
// v tile brought in with GLOBAL_LOAD_ASYNC_TO_LDS_B128 (ASYNCcnt), then
// transposed inside LDS so B-fragments are contiguous ds reads.
// ---------------------------------------------------------------------------
__global__ void __launch_bounds__(256)
k_agent_gather(const __bf16* __restrict__ S, const __bf16* __restrict__ qkvb,
               float* __restrict__ agf) {
  __shared__ __bf16 raw[32][DH];         // async-load landing: [k][d], 4 KB
  __shared__ __bf16 vT[DH][32];          // transposed tile:    [d][k], 4 KB
  int t = threadIdx.x, lane = t & 31, wave = t >> 5;
  int bh = blockIdx.x;
  int g = bh & (H - 1), b = bh >> 4;
  int kbase = blockIdx.y * (N / 8);      // 512-wide K chunk
  int m0 = wave * 16;
  const __bf16* Srow  = S + ((size_t)bh * M + m0) * N;
  const __bf16* Vbase = qkvb + (size_t)b * N * D3 + 2 * D + g * DH;
  int row  = t >> 3;                     // 0..31  (k within tile)
  int col8 = (t & 7) * 8;                // d chunk (8 halves = 16 B)
  unsigned ldsAddr = (unsigned)(uintptr_t)(void*)&raw[row][col8];
  v8f acc[4] = {};
  for (int k0 = kbase; k0 < kbase + N / 8; k0 += 32) {
    unsigned long long ga =
        (unsigned long long)(const void*)(Vbase + (size_t)(k0 + row) * D3 + col8);
    // each of 256 threads moves one 16B chunk of the 32x64 v tile -> LDS
    asm volatile("global_load_async_to_lds_b128 %0, %1, off"
                 :: "v"(ldsAddr), "v"(ga) : "memory");
    asm volatile("s_wait_asynccnt 0x0" ::: "memory");
    __syncthreads();
    {  // transpose raw[k][d] -> vT[d][k]
      v8bf chunk = *(const v8bf*)&raw[row][col8];
#pragma unroll
      for (int e = 0; e < 8; ++e) vT[col8 + e][row] = chunk[e];
    }
    __syncthreads();
    v16bf a = loadA(Srow + k0, N, lane);
#pragma unroll
    for (int j = 0; j < 4; ++j) {
      v16bf bf = loadB(&vT[j * 16][0], 32, lane);   // ds reads
      acc[j] = wmma_bf16(a, bf, acc[j]);
    }
    __syncthreads();
  }
  float* dst = agf + (size_t)bh * DH * M;
#pragma unroll
  for (int j = 0; j < 4; ++j)
#pragma unroll
    for (int e = 0; e < 8; ++e) {
      int m = m0 + e + (lane >> 4) * 8;
      int d = j * 16 + (lane & 15);
      atomicAdd(&dst[(size_t)d * M + m], acc[j][e]);
    }
}

// K7: agf (f32, transposed [b,g,d,m]) -> bf16 for the final WMMA.
__global__ void __launch_bounds__(256)
k_agcvt(const float* __restrict__ agf, __bf16* __restrict__ agbT) {
  int i = blockIdx.x * 256 + threadIdx.x;
  if (i < B * H * DH * M) agbT[i] = (__bf16)agf[i];
}

// ---------------------------------------------------------------------------
// K8: qa scores  Sq[b,h,n,m] = q[b,n,h,:] . a[h,m,:]   (overwrites S buffer)
// ---------------------------------------------------------------------------
__global__ void __launch_bounds__(256)
k_scores_qa(const __bf16* __restrict__ qkvb, const __bf16* __restrict__ ab,
            __bf16* __restrict__ Sq) {
  int lane = threadIdx.x & 31, wave = threadIdx.x >> 5;
  int wid = blockIdx.x * 8 + wave;       // B*H*(N/16)*(M/64) = 32768
  int mhalf = wid & 1;
  int n16   = (wid >> 1) & 255;
  int h     = (wid >> 9) & 15;
  int b     = wid >> 13;
  int n0 = n16 * 16, m0 = mhalf * 64;
  const __bf16* Arow  = qkvb + (size_t)(b * N + n0) * D3 + h * DH;  // q block
  const __bf16* Bbase = ab + (size_t)h * M * DH;
  v8f acc[4] = {};
  for (int k = 0; k < DH; k += 32) {
    v16bf a = loadA(Arow + k, D3, lane);
#pragma unroll
    for (int j = 0; j < 4; ++j) {
      v16bf bf = loadB(Bbase + (size_t)(m0 + j * 16) * DH + k, DH, lane);
      acc[j] = wmma_bf16(a, bf, acc[j]);
    }
  }
  __bf16* dst = Sq + ((size_t)(b * H + h) * N + n0) * M;
#pragma unroll
  for (int j = 0; j < 4; ++j)
#pragma unroll
    for (int e = 0; e < 8; ++e) {
      int nn = e + (lane >> 4) * 8;
      int m  = m0 + j * 16 + (lane & 15);
      dst[(size_t)nn * M + m] = (__bf16)acc[j][e];
    }
}

// ---------------------------------------------------------------------------
// K9: out[b,g,n,d] = mixed_qa[b,g,n,:] @ agbT[b,g,:,:]^T   (K=M=128)
// All 8 waves of a block share agbT[b,g] (contiguous 16 KB) -> one TDM
// tensor_load_to_lds per block; B-fragments become ds reads.
// ---------------------------------------------------------------------------
__global__ void __launch_bounds__(256)
k_out_gemm(const __bf16* __restrict__ Sq, const __bf16* __restrict__ agbT,
           float* __restrict__ out) {
  __shared__ __bf16 agLDS[DH * M];       // 16 KB, [d][m] (K-contiguous cols)
  int lane = threadIdx.x & 31, wave = threadIdx.x >> 5;
  int wid0 = blockIdx.x * 8;             // 8 waves share (b, h)
  int h = (wid0 >> 8) & 15;
  int b = wid0 >> 12;
  int n0 = ((wid0 + wave) & 255) * 16;
  const __bf16* Bglob = agbT + (size_t)(b * H + h) * DH * M;

  if (wave == 0) {
    unsigned long long ga = (unsigned long long)(const void*)Bglob;
    unsigned la = (unsigned)(uintptr_t)(void*)agLDS;
    constexpr unsigned ELEMS = (unsigned)(DH * M);      // 8192 elems, 2B each
    v4u g0;
    g0.x = 1u;                                          // count=1, user mode
    g0.y = la;                                          // lds_addr
    g0.z = (unsigned)(ga & 0xFFFFFFFFu);                // global_addr[31:0]
    g0.w = (unsigned)((ga >> 32) & 0x01FFFFFFu)         // global_addr[56:32]
         | (2u << 30);                                  // type = 2 (image)
    v8i32 g1;
    g1[0] = (int)(1u << 16);                            // data_size = 2 bytes
    g1[1] = (int)((ELEMS & 0xFFFFu) << 16);             // tensor_dim0 lo
    g1[2] = (int)((ELEMS >> 16) | (1u << 16));          // dim0 hi | tensor_dim1=1
    g1[3] = (int)(ELEMS << 16);                         // tile_dim0 = 8192
    g1[4] = 1;                                          // tile_dim1 = 1
    g1[5] = (int)ELEMS;                                 // tensor_dim0_stride
    g1[6] = 0;
    g1[7] = 0;
    tdm_load_to_lds(g0, g1);
    __builtin_amdgcn_s_wait_tensorcnt(0);
  }
  __syncthreads();

  const __bf16* Arow = Sq + ((size_t)(b * H + h) * N + n0) * M;
  v8f acc[4] = {};
  for (int k = 0; k < M; k += 32) {
    v16bf a = loadA(Arow + k, M, lane);
#pragma unroll
    for (int j = 0; j < 4; ++j) {
      v16bf bf = loadB(agLDS + (size_t)(j * 16) * M + k, M, lane);  // ds reads
      acc[j] = wmma_bf16(a, bf, acc[j]);
    }
  }
  float* obase = out + ((size_t)(b * H + h) * N + n0) * DH;
#pragma unroll
  for (int j = 0; j < 4; ++j)
#pragma unroll
    for (int e = 0; e < 8; ++e) {
      int nn = e + (lane >> 4) * 8;
      int d  = j * 16 + (lane & 15);
      obase[(size_t)nn * DH + d] = acc[j][e];
    }
}

// ---------------------------------------------------------------------------
extern "C" void kernel_launch(void* const* d_in, const int* in_sizes, int n_in,
                              void* d_out, int out_size, void* d_ws, size_t ws_size,
                              hipStream_t stream) {
  (void)in_sizes; (void)n_in; (void)out_size; (void)ws_size;
  const float* x      = (const float*)d_in[0];
  const float* agents = (const float*)d_in[1];
  const float* w_fus  = (const float*)d_in[2];
  const float* w_qa   = (const float*)d_in[3];
  const float* w_ak   = (const float*)d_in[4];
  float* out = (float*)d_out;

  char* ws = (char*)d_ws;
  size_t off = 0;
  auto alloc = [&](size_t bytes) -> void* {
    off = (off + 255) & ~(size_t)255;
    void* p = ws + off;
    off += bytes;
    return p;
  };
  __bf16* xb    = (__bf16*)alloc((size_t)BN * D * 2);        // 32 MB
  __bf16* wb    = (__bf16*)alloc((size_t)D3 * D * 2);        // 6 MB
  __bf16* ab    = (__bf16*)alloc((size_t)H * M * DH * 2);    // 0.25 MB
  __bf16* qkvb  = (__bf16*)alloc((size_t)BN * D3 * 2);       // 96 MB
  __bf16* Sbuf  = (__bf16*)alloc((size_t)B * H * M * N * 2); // 64 MB (ak then qa)
  float*  statsA = (float*)alloc((size_t)B * H * M * 2 * 4);
  float*  statsQ = (float*)alloc((size_t)B * H * N * 2 * 4); // 2 MB
  float*  agf    = (float*)alloc((size_t)B * H * DH * M * 4);
  __bf16* agbT   = (__bf16*)alloc((size_t)B * H * DH * M * 2);

  // 1. convert inputs to bf16 (+ zero agent-gather accumulator)
  k_convert<<<4096, 256, 0, stream>>>(x, w_fus, agents, xb, wb, ab, agf);
  // 2. fused QKV projection (dominant GEMM)
  k_qkv<<<dim3(D3 / 64, BN / 128), 256, 0, stream>>>(xb, wb, qkvb);
  // 3..6. ak path: scores -> softmax stats -> in-place talking-heads mix ->
  //        agent_gathered (transposed, bf16)
  k_scores_ak<<<4096, 256, 0, stream>>>(ab, qkvb, Sbuf);
  k_rowstats<<<B * H * M, 256, 0, stream>>>(Sbuf, statsA, N);
  k_mix<<<B * M, 256, 0, stream>>>(Sbuf, statsA, w_ak, M, N);
  k_agent_gather<<<dim3(B * H, 8), 256, 0, stream>>>(Sbuf, qkvb, agf);
  k_agcvt<<<(B * H * DH * M + 255) / 256, 256, 0, stream>>>(agf, agbT);
  // 7..9. qa path (reuses Sbuf): scores -> stats -> mix -> final GEMM to out
  k_scores_qa<<<4096, 256, 0, stream>>>(qkvb, ab, Sbuf);
  k_rowstats<<<B * H * N, 256, 0, stream>>>(Sbuf, statsQ, M);
  k_mix<<<B * N, 256, 0, stream>>>(Sbuf, statsQ, w_qa, N, M);
  k_out_gemm<<<B * H * (N / 16) / 8, 256, 0, stream>>>(Sbuf, agbT, out);
}
